// MultiHeadSelfAttention_75960791598021
// MI455X (gfx1250) — compile-verified, matrix operand order checked by reading, not run on hardware
//
#include <hip/hip_runtime.h>

typedef __attribute__((ext_vector_type(16))) _Float16 v16h;
typedef __attribute__((ext_vector_type(8)))  _Float16 v8h;
typedef __attribute__((ext_vector_type(4)))  _Float16 v4h;
typedef __attribute__((ext_vector_type(8)))  float    v8f;
typedef __attribute__((ext_vector_type(4)))  float    v4f;
typedef __attribute__((ext_vector_type(4)))  int      v4i;

#define NB 4
#define NS 2048
#define ND 1024
#define NH 16
#define NDK 64

// ---------------------------------------------------------------------------
// CDNA5 async global->LDS copy (ASYNCcnt-tracked DMA path).
// ROCm clang signature: (v4i AS1*, v4i AS3*, imm offset, imm cpol).
// ---------------------------------------------------------------------------
#if defined(__has_builtin)
#if __has_builtin(__builtin_amdgcn_global_load_async_to_lds_b128)
#define HAVE_ASYNC_LDS 1
#endif
#if __has_builtin(__builtin_amdgcn_s_wait_asynccnt)
#define HAVE_WAIT_ASYNC 1
#endif
#endif

typedef __attribute__((address_space(1))) v4i gv4i;
typedef __attribute__((address_space(3))) v4i lv4i;

__device__ __forceinline__ void async_b128(const _Float16* g, _Float16* l) {
#if defined(HAVE_ASYNC_LDS)
  __builtin_amdgcn_global_load_async_to_lds_b128(
      (gv4i*)(uintptr_t)g, (lv4i*)(uint32_t)(uintptr_t)l, 0, 0);
#else
  *(v8h*)l = *(const v8h*)g;
#endif
}

__device__ __forceinline__ void wait_async0() {
#if defined(HAVE_ASYNC_LDS)
#if defined(HAVE_WAIT_ASYNC)
  __builtin_amdgcn_s_wait_asynccnt(0);
#else
  asm volatile("s_wait_asynccnt 0x0" ::: "memory");
#endif
#endif
}

// ---------------------------------------------------------------------------
// WMMA wrapper: D(16x16,f32) = A(16x32,f16) * B(32x16,f16) + C
// ---------------------------------------------------------------------------
__device__ __forceinline__ v8f wmma_f16(v16h a, v16h b, v8f c) {
  return __builtin_amdgcn_wmma_f32_16x16x32_f16(
      /*neg_a=*/false, a, /*neg_b=*/false, b,
      /*c_mod=*/(short)0, c, /*reuse_a=*/false, /*reuse_b=*/false);
}

// Fragment loader for A (rows of X / Q / P) and B (rows of W^T / K / Vt).
// CDNA5 16-bit 16x32 layout: lane L (m = L%16, g = L/16) holds
//   VGPR0-3 : K = g*8 + 0..7   (contiguous 16 bytes)
//   VGPR4-7 : K = g*8 + 16..23 (contiguous 16 bytes)
__device__ __forceinline__ v16h load_frag(const _Float16* rowBase, int strideH, int k0) {
  const int lane = threadIdx.x & 31;
  const int g = lane >> 4;
  const _Float16* p = rowBase + (size_t)(lane & 15) * (size_t)strideH + (k0 + g * 8);
  v8h c0 = *(const v8h*)(p);
  v8h c1 = *(const v8h*)(p + 16);
  v16h r;
#pragma unroll
  for (int i = 0; i < 8; ++i) { r[i] = c0[i]; r[i + 8] = c1[i]; }
  return r;
}

// ---------------------------------------------------------------------------
// fp32 -> f16 converters
// ---------------------------------------------------------------------------
__global__ __launch_bounds__(256) void k_cvt_x(const float* __restrict__ x,
                                               _Float16* __restrict__ xh) {
  const size_t i = (size_t)blockIdx.x * 256 + threadIdx.x;
  v4f v = ((const v4f*)x)[i];
  v4h h;
#pragma unroll
  for (int j = 0; j < 4; ++j) h[j] = (_Float16)v[j];
  ((v4h*)xh)[i] = h;
}

__global__ __launch_bounds__(256) void k_cvt_w(
    const float* __restrict__ wq, const float* __restrict__ wk,
    const float* __restrict__ wv, const float* __restrict__ wo,
    _Float16* __restrict__ wqT, _Float16* __restrict__ wkT,
    _Float16* __restrict__ wvT, _Float16* __restrict__ woT) {
  const int f = blockIdx.x * 256 + threadIdx.x;
  const int n = f >> 10;
  const int k = f & (ND - 1);
  const int src = k * ND + n;
  wqT[f] = (_Float16)wq[src];
  wkT[f] = (_Float16)wk[src];
  wvT[f] = (_Float16)wv[src];
  woT[f] = (_Float16)wo[src];
}

// ---------------------------------------------------------------------------
// Tiled WMMA GEMM: out[m][n] = sum_k A[m][k] * Wt[n][k]
// 128x128 tile / 256-thread WG, K-step 32, async-staged LDS.
//   mode 0/1 : RoPE, store f16 to [bh][s][dk]   (Q / K)
//   mode 2   : store f16 transposed [bh][dk][s] (V)
//   mode 3   : store f32 to [m][n]              (output projection)
// ---------------------------------------------------------------------------
__global__ __launch_bounds__(256) void k_gemm(
    const _Float16* __restrict__ A, const _Float16* __restrict__ Wt,
    const long long* __restrict__ tok,
    _Float16* __restrict__ out16, float* __restrict__ out32, int mode) {
  __shared__ __align__(16) _Float16 Xs[128 * 40];  // 40-half row pad
  __shared__ __align__(16) _Float16 Ws[128 * 40];

  const int tid  = threadIdx.x;
  const int lane = tid & 31;
  const int wave = tid >> 5;
  const int wr = wave >> 2;  // 0..1 : 64-row band
  const int wc = wave & 3;   // 0..3 : 32-col band
  const int mBase = blockIdx.x * 128;
  const int nBase = blockIdx.y * 128;

  // staging assignment: 2 16B chunks per thread per tile
  const int srow = tid >> 2;        // 0..63
  const int scol = (tid & 3) * 8;   // 0,8,16,24

  const v8f vz = {0.f, 0.f, 0.f, 0.f, 0.f, 0.f, 0.f, 0.f};
  v8f acc[8];
#pragma unroll
  for (int i = 0; i < 8; ++i) acc[i] = vz;

  for (int k0 = 0; k0 < ND; k0 += 32) {
    async_b128(&A[(size_t)(mBase + srow) * ND + (k0 + scol)],
               &Xs[srow * 40 + scol]);
    async_b128(&A[(size_t)(mBase + srow + 64) * ND + (k0 + scol)],
               &Xs[(srow + 64) * 40 + scol]);
    async_b128(&Wt[(size_t)(nBase + srow) * ND + (k0 + scol)],
               &Ws[srow * 40 + scol]);
    async_b128(&Wt[(size_t)(nBase + srow + 64) * ND + (k0 + scol)],
               &Ws[(srow + 64) * 40 + scol]);
    if (k0 + 32 < ND) {  // pull next K tile toward the WGP
      __builtin_prefetch(&A[(size_t)(mBase + srow) * ND + (k0 + 32)], 0, 3);
      __builtin_prefetch(&Wt[(size_t)(nBase + srow) * ND + (k0 + 32)], 0, 3);
    }
    wait_async0();
    __syncthreads();

    v16h bf0 = load_frag(&Ws[(wc * 32 + 0) * 40], 40, 0);
    v16h bf1 = load_frag(&Ws[(wc * 32 + 16) * 40], 40, 0);
#pragma unroll
    for (int i = 0; i < 4; ++i) {
      v16h af = load_frag(&Xs[(wr * 64 + i * 16) * 40], 40, 0);
      acc[i * 2 + 0] = wmma_f16(af, bf0, acc[i * 2 + 0]);
      acc[i * 2 + 1] = wmma_f16(af, bf1, acc[i * 2 + 1]);
    }
    __syncthreads();
  }

  // ---------------- epilogue ----------------
  // C layout: lane L holds column n = L%16; vector elem r holds row m = r + 8*(L/16)
  const int g  = lane >> 4;
  const int nl = lane & 15;
#pragma unroll
  for (int i = 0; i < 4; ++i) {
    int mRow[8];
    float pos[8];
#pragma unroll
    for (int r = 0; r < 8; ++r) {
      mRow[r] = mBase + wr * 64 + i * 16 + r + 8 * g;
      pos[r] = (mode <= 1) ? (float)tok[mRow[r]] : 0.f;
    }
#pragma unroll
    for (int j = 0; j < 2; ++j) {
      v8f a = acc[i * 2 + j];
      const int n = nBase + wc * 32 + j * 16 + nl;
      const int h = n >> 6;
      const int d = n & (NDK - 1);
      if (mode <= 1) {
        // RoPE: pair partner lives in the adjacent lane
        const float freq = __expf(-0.2878231366242557f * (float)(d >> 1));
        const bool odd = (d & 1);
#pragma unroll
        for (int r = 0; r < 8; ++r) {
          float sv, cv;
          __sincosf(pos[r] * freq, &sv, &cv);
          const float other = __shfl_xor(a[r], 1, 32);
          const float val =
              odd ? (other * sv + a[r] * cv) : (a[r] * cv - other * sv);
          const int bb = mRow[r] >> 11;
          const int sp = mRow[r] & (NS - 1);
          out16[((size_t)(bb * NH + h) * NS + sp) * NDK + d] = (_Float16)val;
        }
      } else if (mode == 2) {  // V stored transposed [bh][dk][s]
#pragma unroll
        for (int r = 0; r < 8; ++r) {
          const int bb = mRow[r] >> 11;
          const int sp = mRow[r] & (NS - 1);
          out16[((size_t)(bb * NH + h) * NDK + d) * NS + sp] = (_Float16)a[r];
        }
      } else {  // fp32 output projection
#pragma unroll
        for (int r = 0; r < 8; ++r) {
          out32[(size_t)mRow[r] * ND + n] = a[r];
        }
      }
    }
  }
}

// ---------------------------------------------------------------------------
// Flash attention: one wave per 16-query tile, stream 32 keys per step.
// Q,K: [bh][s][dk] f16.  Vt: [bh][dk][s] f16.  O: [b][s][D] f16.
// ---------------------------------------------------------------------------
__global__ __launch_bounds__(128) void k_attn(
    const _Float16* __restrict__ Q, const _Float16* __restrict__ K,
    const _Float16* __restrict__ Vt, _Float16* __restrict__ O) {
  __shared__ __align__(16) _Float16 pbuf[4][16 * 32];  // per-wave P tile

  const int tid  = threadIdx.x;
  const int lane = tid & 31;
  const int wave = tid >> 5;
  const int t  = blockIdx.x * 4 + wave;
  const int bh = t >> 7;
  const int qt = t & 127;
  const int g  = lane >> 4;
  const int nl = lane & 15;

  const _Float16* qp = Q + ((size_t)bh * NS + qt * 16) * NDK;
  const _Float16* kp = K + (size_t)bh * NS * NDK;
  const _Float16* vp = Vt + (size_t)bh * NDK * NS;

  const v16h aq0 = load_frag(qp, NDK, 0);
  const v16h aq1 = load_frag(qp, NDK, 32);

  const v8f vz = {0.f, 0.f, 0.f, 0.f, 0.f, 0.f, 0.f, 0.f};
  v8f o[4];
#pragma unroll
  for (int i = 0; i < 4; ++i) o[i] = vz;
  float mr[8], lr[8];
#pragma unroll
  for (int r = 0; r < 8; ++r) { mr[r] = -1e30f; lr[r] = 0.f; }

  _Float16* pb = pbuf[wave];

  for (int ks = 0; ks < NS; ks += 32) {
    if (ks + 32 < NS) {
      __builtin_prefetch(kp + (size_t)(ks + 32 + (lane >> 1)) * NDK, 0, 3);
    }
    // ---- scores: S(16x32) = Q(16x64) . K^T ----
    v8f s0 = vz, s1 = vz;
    {
      const _Float16* kr0 = kp + (size_t)ks * NDK;
      const _Float16* kr1 = kp + (size_t)(ks + 16) * NDK;
      v16h b00 = load_frag(kr0, NDK, 0);
      v16h b01 = load_frag(kr0, NDK, 32);
      s0 = wmma_f16(aq0, b00, s0);
      s0 = wmma_f16(aq1, b01, s0);
      v16h b10 = load_frag(kr1, NDK, 0);
      v16h b11 = load_frag(kr1, NDK, 32);
      s1 = wmma_f16(aq0, b10, s1);
      s1 = wmma_f16(aq1, b11, s1);
    }

    // ---- online softmax (row reductions across 16-lane half-groups) ----
    float p0[8], p1[8], rs[8];
#pragma unroll
    for (int r = 0; r < 8; ++r) {
      const float x0 = s0[r] * 0.125f;  // 1/sqrt(DK)
      const float x1 = s1[r] * 0.125f;
      float tmax = fmaxf(x0, x1);
#pragma unroll
      for (int msk = 1; msk < 16; msk <<= 1)
        tmax = fmaxf(tmax, __shfl_xor(tmax, msk, 32));
      const float mn = fmaxf(mr[r], tmax);
      rs[r] = __expf(mr[r] - mn);
      mr[r] = mn;
      p0[r] = __expf(x0 - mn);
      p1[r] = __expf(x1 - mn);
      float u = p0[r] + p1[r];
#pragma unroll
      for (int msk = 1; msk < 16; msk <<= 1)
        u += __shfl_xor(u, msk, 32);
      lr[r] = lr[r] * rs[r] + u;
    }
#pragma unroll
    for (int t3 = 0; t3 < 4; ++t3)
#pragma unroll
      for (int r = 0; r < 8; ++r) o[t3][r] *= rs[r];

    // ---- transpose P (C-layout -> A-layout) through LDS ----
#pragma unroll
    for (int r = 0; r < 8; ++r) {
      const int m = r + 8 * g;
      pb[m * 32 + nl]      = (_Float16)p0[r];
      pb[m * 32 + 16 + nl] = (_Float16)p1[r];
    }
    asm volatile("s_wait_dscnt 0" ::: "memory");
    const v16h ap = load_frag(pb, 32, 0);

    // ---- O(16x64) += P(16x32) . V(32x64) ----
#pragma unroll
    for (int t3 = 0; t3 < 4; ++t3) {
      v16h bv = load_frag(vp + (size_t)(t3 * 16) * NS, NS, ks);
      o[t3] = wmma_f16(ap, bv, o[t3]);
    }
  }

  // ---- finalize ----
  float inv[8];
#pragma unroll
  for (int r = 0; r < 8; ++r) inv[r] = 1.0f / lr[r];
  const int bb = bh >> 4;
  const int hh = bh & 15;
#pragma unroll
  for (int t3 = 0; t3 < 4; ++t3) {
#pragma unroll
    for (int r = 0; r < 8; ++r) {
      const int srow = qt * 16 + r + 8 * g;
      O[((size_t)bb * NS + srow) * ND + hh * NDK + t3 * 16 + nl] =
          (_Float16)(o[t3][r] * inv[r]);
    }
  }
}

// ---------------------------------------------------------------------------
// Host side
// ---------------------------------------------------------------------------
extern "C" void kernel_launch(void* const* d_in, const int* in_sizes, int n_in,
                              void* d_out, int out_size, void* d_ws, size_t ws_size,
                              hipStream_t stream) {
  const float*     x   = (const float*)d_in[0];
  const long long* tok = (const long long*)d_in[1];
  const float*     Wq  = (const float*)d_in[2];
  const float*     Wk  = (const float*)d_in[3];
  const float*     Wv  = (const float*)d_in[4];
  const float*     Wo  = (const float*)d_in[5];
  float* out = (float*)d_out;

  const size_t XHN = (size_t)NB * NS * ND;
  const size_t WTN = (size_t)ND * ND;
  const size_t QKN = (size_t)NB * NH * NS * NDK;

  _Float16* xh  = (_Float16*)d_ws;
  _Float16* wqT = xh + XHN;
  _Float16* wkT = wqT + WTN;
  _Float16* wvT = wkT + WTN;
  _Float16* woT = wvT + WTN;
  _Float16* Qw  = woT + WTN;
  _Float16* Kw  = Qw + QKN;
  _Float16* Vw  = Kw + QKN;
  _Float16* AO  = Vw + QKN;
  (void)in_sizes; (void)n_in; (void)out_size; (void)ws_size;

  k_cvt_x<<<dim3(XHN / (256 * 4)), 256, 0, stream>>>(x, xh);
  k_cvt_w<<<dim3(WTN / 256), 256, 0, stream>>>(Wq, Wk, Wv, Wo, wqT, wkT, wvT, woT);

  dim3 gg(NB * NS / 128, ND / 128);  // 64 x 8
  k_gemm<<<gg, 256, 0, stream>>>(xh, wqT, tok, Qw, nullptr, 0);
  k_gemm<<<gg, 256, 0, stream>>>(xh, wkT, tok, Kw, nullptr, 1);
  k_gemm<<<gg, 256, 0, stream>>>(xh, wvT, tok, Vw, nullptr, 2);

  k_attn<<<dim3(NB * NH * (NS / 16) / 4), 128, 0, stream>>>(Qw, Kw, Vw, AO);

  k_gemm<<<gg, 256, 0, stream>>>(AO, woT, tok, nullptr, out, 3);
}
